// SwinTransformerBlock_17540646437653
// MI455X (gfx1250) — compile-verified, matrix operand order checked by reading, not run on hardware
//
#include <hip/hip_runtime.h>
#include <hip/hip_bf16.h>
#include <math.h>

// ---------------------------------------------------------------------------
// Swin Transformer block, fused per-window kernel for gfx1250 (MI455X).
//  - bf16 WMMA (v_wmma_f32_16x16x32_bf16) for all matmuls, f32 accumulation
//  - TDM (tensor_load_to_lds / tensor_store_from_lds) for the strided
//    NCHW 7x7x96 window gather/scatter, tracked with TENSORcnt
//  - LN / softmax / GELU in f32.  One workgroup (8 wave32s) per window.
// ---------------------------------------------------------------------------

#define USE_TDM 1

typedef __attribute__((ext_vector_type(16))) __bf16 v16bf;
typedef __attribute__((ext_vector_type(8)))  float  v8f;
typedef __attribute__((ext_vector_type(4)))  unsigned int v4u;
typedef __attribute__((ext_vector_type(8)))  unsigned int v8u;

#define DIM   96
#define HH    224
#define WW    224
#define NTOK  49     // 7*7 tokens per window

// LDS region byte offsets (total 143744 B -> 2 workgroups per 320KB WGP)
#define OFF_XC    0                       // float [96][49]  x window (c-major), residual y1
#define OFF_XA    18816                   // u16   [64][104] bf16 activations
#define OFF_ACT   32128                   // u16   [64][392] qkv / h1 (P overlays)
#define OFF_SC    82304                   // float [3][64][64] scores; outb_c overlays
#define OFF_VT    131456                  // u16   [3*32][64] V transposed
#define SMEM_BYTES 143744

__device__ __forceinline__ unsigned short f2bf(float f) {
  unsigned int u = __float_as_uint(f);
  u += 0x7FFFu + ((u >> 16) & 1u);          // round-to-nearest-even
  return (unsigned short)(u >> 16);
}

__device__ __forceinline__ unsigned rfl(unsigned v) {
  return (unsigned)__builtin_amdgcn_readfirstlane((int)v);
}

union Frag { v16bf v; unsigned int u[8]; };

// Load a 16(rows) x 32(k) bf16 fragment from a row-major u16 matrix.
// CDNA5 ISA 16-bit A-matrix layout:
//   lane L: row = row0 + (L&15)
//   halfword h: k = k0 + 16*(h>>3) + 8*(L>>4) + (h&7)
// Used for A operands and for B operands given a transposed (N-major) source.
__device__ __forceinline__ v16bf ld_frag(const unsigned short* __restrict__ base,
                                         int row0, int stride, int k0, int lane) {
  Frag f;
  const unsigned short* p = base + (row0 + (lane & 15)) * stride
                                 + k0 + ((lane >> 4) << 3);
  const unsigned int* q0 = (const unsigned int*)p;
  const unsigned int* q1 = (const unsigned int*)(p + 16);
  f.u[0] = q0[0]; f.u[1] = q0[1]; f.u[2] = q0[2]; f.u[3] = q0[3];
  f.u[4] = q1[0]; f.u[5] = q1[1]; f.u[6] = q1[2]; f.u[7] = q1[3];
  return f.v;
}

__device__ __forceinline__ v8f wmma_bf16(v16bf a, v16bf b, v8f c) {
  return __builtin_amdgcn_wmma_f32_16x16x32_bf16(false, a, false, b,
                                                 (short)0, c, false, false);
}

#if USE_TDM
// ---------------------------------------------------------------------------
// Tensor Data Mover descriptor (D#), per CDNA5 ISA.
// 3D tile: x = 7 f32 (contiguous), y = 7 rows (stride 224), z = 96 planes
// (stride 224*224).  data_size = 4B.  global_addr points at the tile start.
// ---------------------------------------------------------------------------
struct TdmDesc { v4u g0; v8u g1; v4u g2; v4u g3; };

__device__ __forceinline__ TdmDesc make_window_desc(unsigned lds_off,
                                                    unsigned long long ga) {
  TdmDesc d;
  // group0: count=1 | lds_addr | global_addr[56:0] | type=2 (bits 127:126)
  d.g0.x = rfl(1u);
  d.g0.y = rfl(lds_off);
  d.g0.z = rfl((unsigned)ga);
  d.g0.w = rfl(((unsigned)(ga >> 32) & 0x01FFFFFFu) | 0x80000000u);
  // group1
  d.g1.s0 = rfl(0x00020000u);          // wg_mask=0, data_size=2 (4B), no flags
  d.g1.s1 = rfl(224u << 16);           // tensor_dim0[15:0] @ bits 63:48
  d.g1.s2 = rfl(224u << 16);           // dim0[31:16]=0 | tensor_dim1[15:0]
  d.g1.s3 = rfl(7u << 16);             // dim1[31:16]=0 | tile_dim0=7
  d.g1.s4 = rfl(7u | (96u << 16));     // tile_dim1=7 | tile_dim2=96
  d.g1.s5 = rfl(224u);                 // tensor_dim0_stride[31:0]
  d.g1.s6 = rfl(50176u << 16);         // stride0[47:32]=0 | stride1[15:0]
  d.g1.s7 = rfl(0u);                   // stride1[47:16] = 0
  // group2: tensor_dim2=96, rest 0 (3D tile, no iteration)
  d.g2.x = rfl(96u); d.g2.y = rfl(0u); d.g2.z = rfl(0u); d.g2.w = rfl(0u);
  // group3: zeros
  d.g3.x = rfl(0u); d.g3.y = rfl(0u); d.g3.z = rfl(0u); d.g3.w = rfl(0u);
  return d;
}

__device__ __forceinline__ void tdm_load(const TdmDesc& d) {
  asm volatile("tensor_load_to_lds %0, %1, %2, %3"
               :: "s"(d.g0), "s"(d.g1), "s"(d.g2), "s"(d.g3) : "memory");
}
__device__ __forceinline__ void tdm_store(const TdmDesc& d) {
  asm volatile("tensor_store_from_lds %0, %1, %2, %3"
               :: "s"(d.g0), "s"(d.g1), "s"(d.g2), "s"(d.g3) : "memory");
}
#endif

// ---------------------------------------------------------------------------
// Kernel 0: convert weights f32 -> bf16, transposed to N-major in workspace.
// ---------------------------------------------------------------------------
__global__ __launch_bounds__(256) void prep_weights_kernel(
    const float* __restrict__ qkv_w, const float* __restrict__ proj_w,
    const float* __restrict__ mlp_w1, const float* __restrict__ mlp_w2,
    unsigned short* __restrict__ ws) {
  int i = blockIdx.x * 256 + threadIdx.x;
  if (i < 27648) {                       // qkv_w (96,288) -> [n=288][k=96]
    int n = i / 96, k = i % 96;
    ws[i] = f2bf(qkv_w[k * 288 + n]);
    return;
  }
  i -= 27648;
  if (i < 9216) {                        // proj_w (96,96) -> [n=96][k=96]
    int n = i / 96, k = i % 96;
    ws[27648 + i] = f2bf(proj_w[k * 96 + n]);
    return;
  }
  i -= 9216;
  if (i < 36864) {                       // mlp_w1 (96,384) -> [n=384][k=96]
    int n = i / 96, k = i % 96;
    ws[36864 + i] = f2bf(mlp_w1[k * 384 + n]);
    return;
  }
  i -= 36864;
  if (i < 36864) {                       // mlp_w2 (384,96) -> [n=96][k=384]
    int n = i / 384, k = i % 384;
    ws[73728 + i] = f2bf(mlp_w2[k * 96 + n]);
  }
}

// ---------------------------------------------------------------------------
// Main fused kernel: one workgroup per 7x7 window.
// ---------------------------------------------------------------------------
__global__ __launch_bounds__(256) void swin_block_kernel(
    const float* __restrict__ x,
    const float* __restrict__ ln1_g, const float* __restrict__ ln1_b,
    const float* __restrict__ qkv_b,
    const float* __restrict__ rpb, const int* __restrict__ rel_index,
    const float* __restrict__ proj_b,
    const float* __restrict__ ln2_g, const float* __restrict__ ln2_b,
    const float* __restrict__ mlp_b1, const float* __restrict__ mlp_b2,
    const unsigned short* __restrict__ wq,   // [288][96] bf16 (N-major)
    const unsigned short* __restrict__ wp,   // [96][96]
    const unsigned short* __restrict__ w1,   // [384][96]
    const unsigned short* __restrict__ w2,   // [96][384]
    float* __restrict__ out) {
  __shared__ __align__(16) char smem[SMEM_BYTES];
  float*          xc   = (float*)(smem + OFF_XC);            // [96][49] c-major
  unsigned short* xa   = (unsigned short*)(smem + OFF_XA);   // [64][104]
  unsigned short* act  = (unsigned short*)(smem + OFF_ACT);  // [64][392]
  unsigned short* pmat = act;                                // overlay [3*64][64]
  float*          sc   = (float*)(smem + OFF_SC);            // [3*64][64]
  float*          outc = sc;                                 // overlay [96][49]
  unsigned short* vt   = (unsigned short*)(smem + OFF_VT);   // [3*32][64]

  const int tid  = threadIdx.x;
  const int lane = tid & 31;
  const int wave = tid >> 5;

  const int blk = blockIdx.x;
  const int b   = blk >> 10;          // batch       (16)
  const int wy  = (blk >> 5) & 31;    // window row  (32)
  const int wx  = blk & 31;           // window col  (32)
  const int y0 = wy * 7, x0 = wx * 7;
  const long long base_io = (long long)b * DIM * HH * WW;
  const long long tile_off = base_io + (long long)y0 * WW + x0;

  // ---- Phase 1: DMA window (NCHW f32) into LDS as [c][i][j] ----
#if USE_TDM
  if (wave == 0) {
    TdmDesc dl = make_window_desc((unsigned)(size_t)(void*)xc,
                                  (unsigned long long)(const void*)(x + tile_off));
    tdm_load(dl);
    __builtin_amdgcn_s_wait_tensorcnt(0);
  }
#else
  for (int e = tid; e < NTOK * DIM; e += 256) {
    int j = e % 7, t2 = e / 7, i = t2 % 7, c = t2 / 7;
    xc[c * 49 + i * 7 + j] = x[tile_off + ((long long)c * HH + i) * WW + j];
  }
#endif
  // Early prefetch of weight tables into caches (global_prefetch_b8).
  if (wave == 7) {
    __builtin_prefetch((const char*)wq + lane * 1728, 0, 1);
    __builtin_prefetch((const char*)w1 + lane * 2304, 0, 1);
    __builtin_prefetch((const char*)w2 + lane * 2304, 0, 1);
  }
  __syncthreads();

  // ---- Phase 2: LayerNorm1 -> xa (bf16 token-major), zero pad rows 49..63 ----
  if (tid < 64) {
    if (tid < NTOK) {
      float mu = 0.f;
      for (int c = 0; c < 96; ++c) mu += xc[c * 49 + tid];
      mu *= (1.f / 96.f);
      float var = 0.f;
      for (int c = 0; c < 96; ++c) { float d = xc[c * 49 + tid] - mu; var += d * d; }
      var *= (1.f / 96.f);
      float rs = rsqrtf(var + 1e-5f);
      for (int c = 0; c < 96; ++c)
        xa[tid * 104 + c] = f2bf((xc[c * 49 + tid] - mu) * rs * ln1_g[c] + ln1_b[c]);
    } else {
      for (int c = 0; c < 96; ++c) xa[tid * 104 + c] = 0;
    }
  }
  __syncthreads();

  // ---- Phase 3: qkv = xn @ Wqkv + b   (64x96 @ 96x288) ----
  for (int nt = wave; nt < 18; nt += 8) {
    int n0 = nt * 16;
    v8f c0 = {0,0,0,0,0,0,0,0}, c1 = {0,0,0,0,0,0,0,0};
    v8f c2 = {0,0,0,0,0,0,0,0}, c3 = {0,0,0,0,0,0,0,0};
    for (int ks = 0; ks < 96; ks += 32) {
      v16bf bf = ld_frag(wq, n0, 96, ks, lane);
      c0 = wmma_bf16(ld_frag(xa,  0, 104, ks, lane), bf, c0);
      c1 = wmma_bf16(ld_frag(xa, 16, 104, ks, lane), bf, c1);
      c2 = wmma_bf16(ld_frag(xa, 32, 104, ks, lane), bf, c2);
      c3 = wmma_bf16(ld_frag(xa, 48, 104, ks, lane), bf, c3);
    }
    int n = n0 + (lane & 15);
    float bb = qkv_b[n];
    int mb = (lane >> 4) << 3;
#pragma unroll
    for (int r = 0; r < 8; ++r) {
      act[(mb + r)      * 392 + n] = f2bf(c0[r] + bb);
      act[(16 + mb + r) * 392 + n] = f2bf(c1[r] + bb);
      act[(32 + mb + r) * 392 + n] = f2bf(c2[r] + bb);
      act[(48 + mb + r) * 392 + n] = f2bf(c3[r] + bb);
    }
  }
  __syncthreads();

  // ---- Phase 4: stage V transposed: vt[h*32+d][token] ----
  for (int e = tid; e < 3 * 32 * 64; e += 256) {
    int t = e & 63, d = (e >> 6) & 31, h = e >> 11;
    vt[(h * 32 + d) * 64 + t] = act[t * 392 + 192 + h * 32 + d];
  }
  __syncthreads();

  // ---- Phase 5: scores = q @ k^T per head (hd=32 -> one WMMA per tile) ----
  for (int job = wave; job < 48; job += 8) {
    int h = job >> 4, mt = (job >> 2) & 3, nt = job & 3;
    int m0 = mt * 16, n0 = nt * 16;
    v8f c = {0,0,0,0,0,0,0,0};
    v16bf a  = ld_frag(act, m0, 392, h * 32, lane);        // q rows
    v16bf bf = ld_frag(act, n0, 392, 96 + h * 32, lane);   // k rows (B col-major)
    c = wmma_bf16(a, bf, c);
    int n = n0 + (lane & 15);
    int mb = m0 + ((lane >> 4) << 3);
#pragma unroll
    for (int r = 0; r < 8; ++r) sc[(h * 64 + mb + r) * 64 + n] = c[r];
  }
  __syncthreads();

  // ---- Phase 6: scale + rel-pos bias + softmax -> pmat (bf16, overlays act) ----
  if (tid < 3 * NTOK) {
    int h = tid / NTOK, m = tid % NTOK;
    float* srow = &sc[(h * 64 + m) * 64];
    const int* ri = &rel_index[m * NTOK];
    const float scale = 0.17677669529663689f;   // 32^-0.5
    float mx = -1e30f;
    for (int n = 0; n < NTOK; ++n) {
      float v = srow[n] * scale + rpb[ri[n] * 3 + h];
      mx = fmaxf(mx, v);
    }
    float sum = 0.f;
    for (int n = 0; n < NTOK; ++n) {
      float e = __expf(srow[n] * scale + rpb[ri[n] * 3 + h] - mx);
      srow[n] = e; sum += e;
    }
    float rs = 1.f / sum;
    unsigned short* prow = &pmat[(h * 64 + m) * 64];
    for (int n = 0; n < NTOK; ++n) prow[n] = f2bf(srow[n] * rs);
    for (int n = NTOK; n < 64; ++n) prow[n] = 0;   // zero padded keys
  }
  __syncthreads();

  // ---- Phase 7: attn_out = P @ V  -> xa (bf16, heads concatenated) ----
  for (int job = wave; job < 24; job += 8) {
    int h = job >> 3, mt = (job >> 1) & 3, nt = job & 1;
    int m0 = mt * 16, n0 = nt * 16;
    v8f c = {0,0,0,0,0,0,0,0};
    for (int ks = 0; ks < 64; ks += 32) {
      v16bf a  = ld_frag(pmat, h * 64 + m0, 64, ks, lane);
      v16bf bf = ld_frag(vt,   h * 32 + n0, 64, ks, lane);
      c = wmma_bf16(a, bf, c);
    }
    int n = n0 + (lane & 15);
    int mb = m0 + ((lane >> 4) << 3);
#pragma unroll
    for (int r = 0; r < 8; ++r)
      xa[(mb + r) * 104 + h * 32 + n] = f2bf(c[r]);
  }
  __syncthreads();

  // ---- Phase 8: proj + residual into xc (channel-major, guarded m<49) ----
  for (int job = wave; job < 12; job += 8) {
    int nt = job >> 1;
    int m0 = (job & 1) * 32;
    int n0 = nt * 16;
    v8f c0 = {0,0,0,0,0,0,0,0}, c1 = {0,0,0,0,0,0,0,0};
    for (int ks = 0; ks < 96; ks += 32) {
      v16bf bf = ld_frag(wp, n0, 96, ks, lane);
      c0 = wmma_bf16(ld_frag(xa, m0,      104, ks, lane), bf, c0);
      c1 = wmma_bf16(ld_frag(xa, m0 + 16, 104, ks, lane), bf, c1);
    }
    int n = n0 + (lane & 15);
    float bb = proj_b[n];
    int mb = m0 + ((lane >> 4) << 3);
#pragma unroll
    for (int r = 0; r < 8; ++r) {
      int m = mb + r;
      if (m < NTOK)      xc[n * 49 + m]      += c0[r] + bb;
      if (m + 16 < NTOK) xc[n * 49 + m + 16] += c1[r] + bb;
    }
  }
  __syncthreads();

  // ---- Phase 9: LayerNorm2 on y1 -> xa (bf16) ----
  if (tid < 64) {
    if (tid < NTOK) {
      float mu = 0.f;
      for (int c = 0; c < 96; ++c) mu += xc[c * 49 + tid];
      mu *= (1.f / 96.f);
      float var = 0.f;
      for (int c = 0; c < 96; ++c) { float d = xc[c * 49 + tid] - mu; var += d * d; }
      var *= (1.f / 96.f);
      float rs = rsqrtf(var + 1e-5f);
      for (int c = 0; c < 96; ++c)
        xa[tid * 104 + c] = f2bf((xc[c * 49 + tid] - mu) * rs * ln2_g[c] + ln2_b[c]);
    } else {
      for (int c = 0; c < 96; ++c) xa[tid * 104 + c] = 0;
    }
  }
  __syncthreads();

  // ---- Phase 10: h1 = GELU(xn2 @ W1 + b1) -> act (bf16) ----
  for (int nt = wave; nt < 24; nt += 8) {
    int n0 = nt * 16;
    v8f c0 = {0,0,0,0,0,0,0,0}, c1 = {0,0,0,0,0,0,0,0};
    v8f c2 = {0,0,0,0,0,0,0,0}, c3 = {0,0,0,0,0,0,0,0};
    for (int ks = 0; ks < 96; ks += 32) {
      v16bf bf = ld_frag(w1, n0, 96, ks, lane);
      c0 = wmma_bf16(ld_frag(xa,  0, 104, ks, lane), bf, c0);
      c1 = wmma_bf16(ld_frag(xa, 16, 104, ks, lane), bf, c1);
      c2 = wmma_bf16(ld_frag(xa, 32, 104, ks, lane), bf, c2);
      c3 = wmma_bf16(ld_frag(xa, 48, 104, ks, lane), bf, c3);
    }
    int n = n0 + (lane & 15);
    float bb = mlp_b1[n];
    int mb = (lane >> 4) << 3;
#pragma unroll
    for (int r = 0; r < 8; ++r) {
      float v0 = c0[r] + bb, v1 = c1[r] + bb, v2 = c2[r] + bb, v3 = c3[r] + bb;
      act[(mb + r)      * 392 + n] = f2bf(0.5f * v0 * (1.f + erff(v0 * 0.70710678f)));
      act[(16 + mb + r) * 392 + n] = f2bf(0.5f * v1 * (1.f + erff(v1 * 0.70710678f)));
      act[(32 + mb + r) * 392 + n] = f2bf(0.5f * v2 * (1.f + erff(v2 * 0.70710678f)));
      act[(48 + mb + r) * 392 + n] = f2bf(0.5f * v3 * (1.f + erff(v3 * 0.70710678f)));
    }
  }
  __syncthreads();

  // ---- Phase 11: out = y1 + h1 @ W2 + b2 -> outc [96][49] (overlays sc) ----
  for (int job = wave; job < 12; job += 8) {
    int nt = job >> 1;
    int m0 = (job & 1) * 32;
    int n0 = nt * 16;
    v8f c0 = {0,0,0,0,0,0,0,0}, c1 = {0,0,0,0,0,0,0,0};
    for (int ks = 0; ks < 384; ks += 32) {
      v16bf bf = ld_frag(w2, n0, 384, ks, lane);
      c0 = wmma_bf16(ld_frag(act, m0,      392, ks, lane), bf, c0);
      c1 = wmma_bf16(ld_frag(act, m0 + 16, 392, ks, lane), bf, c1);
    }
    int n = n0 + (lane & 15);
    float bb = mlp_b2[n];
    int mb = m0 + ((lane >> 4) << 3);
#pragma unroll
    for (int r = 0; r < 8; ++r) {
      int m = mb + r;
      if (m < NTOK)      outc[n * 49 + m]      = xc[n * 49 + m]      + c0[r] + bb;
      if (m + 16 < NTOK) outc[n * 49 + m + 16] = xc[n * 49 + m + 16] + c1[r] + bb;
    }
  }
  __syncthreads();

  // ---- Phase 12: DMA window back to NCHW ----
#if USE_TDM
  if (wave == 0) {
    TdmDesc ds = make_window_desc((unsigned)(size_t)(void*)outc,
                                  (unsigned long long)(const void*)(out + tile_off));
    tdm_store(ds);
    __builtin_amdgcn_s_wait_tensorcnt(0);
  }
#else
  for (int e = tid; e < NTOK * DIM; e += 256) {
    int j = e % 7, t2 = e / 7, i = t2 % 7, c = t2 / 7;
    out[tile_off + ((long long)c * HH + i) * WW + j] = outc[c * 49 + i * 7 + j];
  }
#endif
}

// ---------------------------------------------------------------------------
extern "C" void kernel_launch(void* const* d_in, const int* in_sizes, int n_in,
                              void* d_out, int out_size, void* d_ws, size_t ws_size,
                              hipStream_t stream) {
  (void)in_sizes; (void)n_in; (void)out_size; (void)ws_size;
  const float* x      = (const float*)d_in[0];
  const float* ln1_g  = (const float*)d_in[1];
  const float* ln1_b  = (const float*)d_in[2];
  const float* qkv_w  = (const float*)d_in[3];
  const float* qkv_b  = (const float*)d_in[4];
  const float* rpb    = (const float*)d_in[5];
  const float* proj_w = (const float*)d_in[6];
  const float* proj_b = (const float*)d_in[7];
  const float* ln2_g  = (const float*)d_in[8];
  const float* ln2_b  = (const float*)d_in[9];
  const float* mlp_w1 = (const float*)d_in[10];
  const float* mlp_b1 = (const float*)d_in[11];
  const float* mlp_w2 = (const float*)d_in[12];
  const float* mlp_b2 = (const float*)d_in[13];
  const int*   rel_ix = (const int*)d_in[14];

  unsigned short* wbf = (unsigned short*)d_ws;   // bf16 transposed weights

  prep_weights_kernel<<<432, 256, 0, stream>>>(qkv_w, proj_w, mlp_w1, mlp_w2, wbf);

  swin_block_kernel<<<16384, 256, 0, stream>>>(
      x, ln1_g, ln1_b, qkv_b, rpb, rel_ix, proj_b, ln2_g, ln2_b,
      mlp_b1, mlp_b2,
      wbf,            // qkv_w^T  [288][96]
      wbf + 27648,    // proj_w^T [96][96]
      wbf + 36864,    // mlp_w1^T [384][96]
      wbf + 73728,    // mlp_w2^T [96][384]
      (float*)d_out);
}
